// Block_52218212384827
// MI455X (gfx1250) — matrix operand order repaired and checked by reading, not compiled, not run on hardware
//
#include <hip/hip_runtime.h>
#include <hip/hip_bf16.h>

typedef __attribute__((ext_vector_type(16))) _Float16 v16h;
typedef __attribute__((ext_vector_type(8)))  float    v8f;

union FragA { v16h h; unsigned u[8]; };

#define BB 4
#define TT 2048
#define EE 1024
#define HH 16
#define HS 64
#define MM (BB * TT)          // 8192 rows
#define FFN_VALID 100
#define FFN_PAD 128

// ---------------------------------------------------------------------------
// LayerNorm: one block (256 threads, 8 waves) per row; writes f16 output.
// ---------------------------------------------------------------------------
__global__ __launch_bounds__(256)
void ln_kernel(const float* __restrict__ x, const float* __restrict__ g,
               const float* __restrict__ b, _Float16* __restrict__ out)
{
    const int row = blockIdx.x;
    const float* xr = x + (size_t)row * EE;
    __shared__ float sA[8], sB[8];

    float s = 0.f, q = 0.f;
    for (int i = threadIdx.x; i < EE; i += 256) {
        float v = xr[i];
        s += v; q += v * v;
    }
#pragma unroll
    for (int off = 16; off > 0; off >>= 1) {
        s += __shfl_xor(s, off, 32);
        q += __shfl_xor(q, off, 32);
    }
    const int w = threadIdx.x >> 5;
    if ((threadIdx.x & 31) == 0) { sA[w] = s; sB[w] = q; }
    __syncthreads();
    if (threadIdx.x < 8) {
        float a = sA[threadIdx.x], c = sB[threadIdx.x];
#pragma unroll
        for (int off = 4; off > 0; off >>= 1) {
            a += __shfl_xor(a, off, 32);
            c += __shfl_xor(c, off, 32);
        }
        if (threadIdx.x == 0) { sA[0] = a; sB[0] = c; }
    }
    __syncthreads();
    const float mu  = sA[0] * (1.f / EE);
    const float var = sB[0] * (1.f / EE) - mu * mu;
    const float rstd = rsqrtf(var + 1e-5f);
    for (int i = threadIdx.x; i < EE; i += 256)
        out[(size_t)row * EE + i] = (_Float16)((xr[i] - mu) * rstd * g[i] + b[i]);
}

// ---------------------------------------------------------------------------
// Transpose + f32->f16 convert with zero padding:
//  W [K x N] row-major (f32)  ->  WT [Npad x Kpad] row-major (f16)
// ---------------------------------------------------------------------------
__global__ __launch_bounds__(256)
void wtrans_kernel(const float* __restrict__ W, _Float16* __restrict__ WT,
                   int K, int N, int Npad, int Kpad)
{
    int idx = blockIdx.x * 256 + threadIdx.x;
    if (idx >= Npad * Kpad) return;
    int n = idx / Kpad, k = idx % Kpad;
    float v = (n < N && k < K) ? W[(size_t)k * N + n] : 0.f;
    WT[idx] = (_Float16)v;
}

// ---------------------------------------------------------------------------
// V transpose: V [B*T, E] f16 -> VT [B, H, HS, T] f16 so that consecutive
// keys are contiguous (packed-pair loads for WMMA B fragments).
// ---------------------------------------------------------------------------
__global__ __launch_bounds__(256)
void vtrans_kernel(const _Float16* __restrict__ V, _Float16* __restrict__ VT)
{
    int idx = blockIdx.x * 256 + threadIdx.x;      // over B*T*E, read-coalesced
    if (idx >= MM * EE) return;
    const int e  = idx & (EE - 1);
    const int bt = idx >> 10;                      // b*T + t
    const int t  = bt & (TT - 1);
    const int bb = bt >> 11;
    const int hh = e >> 6;
    const int d  = e & (HS - 1);
    VT[(((size_t)(bb * HH + hh) * HS) + d) * TT + t] = V[idx];
}

// ---------------------------------------------------------------------------
// Generic WMMA GEMM: C = scale * (A[MxK,f16,row] * B), B given as BT[NxK,row].
// One wave computes a 32(M) x 64(N) tile: 8 accumulators, 2 A-frags x 4
// B-frags per k-step (6 dwords of fragment traffic per WMMA).
// Epilogue: optional bias (n<Nvalid), optional f32 residual, optional ReLU;
// OUT_F16 stores f16 (zeros for n>=Nvalid padding), else stores f32.
// ---------------------------------------------------------------------------
template <bool OUT_F16, bool RELU>
__global__ __launch_bounds__(128)
void wmma_gemm_kernel(const _Float16* __restrict__ A,
                      const _Float16* __restrict__ BT,
                      const float* __restrict__ bias,
                      const float* __restrict__ residual,
                      _Float16* __restrict__ Ch, float* __restrict__ Cf,
                      int M, int N, int K, int Nvalid, int ldc, float scale)
{
    const int wave   = blockIdx.x * (blockDim.x >> 5) + (threadIdx.x >> 5);
    const int tilesN = N >> 6;
    const int tilesM = M >> 5;
    if (wave >= tilesM * tilesN) return;
    const int lane  = threadIdx.x & 31;
    const int rm    = wave / tilesN;
    const int cn    = wave % tilesN;
    const int half_ = lane >> 4;
    const int l16   = lane & 15;

    const v8f vzero = {};
    v8f acc[2][4];
#pragma unroll
    for (int g = 0; g < 2; ++g)
#pragma unroll
        for (int t = 0; t < 4; ++t) acc[g][t] = vzero;

    const _Float16* Arow0 = A + (size_t)(rm * 32 + l16) * K;
    const _Float16* Arow1 = Arow0 + (size_t)16 * K;

    for (int kk = 0; kk < K; kk += 32) {
        FragA a0, a1;
#pragma unroll
        for (int j = 0; j < 8; ++j) {
            // 16-bit A 16x32 layout: K = 16*(j/4) + 8*half + 2*(j&3) (+lo)
            const int koff = kk + ((j >> 2) << 4) + (half_ << 3) + ((j & 3) << 1);
            a0.u[j] = *(const unsigned*)(Arow0 + koff);
            a1.u[j] = *(const unsigned*)(Arow1 + koff);
        }
#pragma unroll
        for (int t = 0; t < 4; ++t) {
            const int n = cn * 64 + t * 16 + l16;
            const _Float16* Brow = BT + (size_t)n * K + kk;
            FragA bfr;
#pragma unroll
            for (int j = 0; j < 8; ++j)
                bfr.u[j] = *(const unsigned*)(Brow + ((j >> 2) << 4) + (half_ << 3) + ((j & 3) << 1));
            acc[0][t] = __builtin_amdgcn_wmma_f32_16x16x32_f16(
                false, a0.h, false, bfr.h, (short)0, acc[0][t], false, false);
            acc[1][t] = __builtin_amdgcn_wmma_f32_16x16x32_f16(
                false, a1.h, false, bfr.h, (short)0, acc[1][t], false, false);
        }
    }

#pragma unroll
    for (int g = 0; g < 2; ++g)
#pragma unroll
        for (int t = 0; t < 4; ++t) {
            const int n = cn * 64 + t * 16 + l16;
#pragma unroll
            for (int r = 0; r < 8; ++r) {
                const int m = rm * 32 + g * 16 + half_ * 8 + r;
                float v = acc[g][t][r] * scale;
                if (bias && n < Nvalid) v += bias[n];
                if (residual) v += residual[(size_t)m * ldc + n];
                if (RELU) v = v > 0.f ? v : 0.f;
                if (OUT_F16) {
                    Ch[(size_t)m * ldc + n] =
                        (n < Nvalid) ? (_Float16)v : (_Float16)0.f;
                } else if (n < Nvalid) {
                    Cf[(size_t)m * ldc + n] = v;
                }
            }
        }
}

// ---------------------------------------------------------------------------
// Flash attention: one wave per (b, h, 16-row Q tile).
// 32-key blocks; softmax stats updated ONCE per block (both 16-key sub-tiles
// folded before the 16-lane xor reductions). Causal masking applied only in
// the single diagonal block. Scores arrive pre-scaled (Q GEMM scaled 1/32).
// ---------------------------------------------------------------------------
template <bool MASKED>
__device__ __forceinline__ void attn_block(
    int s0, int qt, int half_, int l16,
    const FragA& aq0, const FragA& aq1,
    const _Float16* __restrict__ kr_base,   // K rows for this (b), head col off
    const _Float16* __restrict__ vt_base,   // VT rows for this (b,h)
    _Float16* __restrict__ pb,              // this wave's 16x32 LDS tile
    float row_max[8], float row_sum[8], v8f o_acc[4])
{
    v8f sc[2];
#pragma unroll
    for (int sub = 0; sub < 2; ++sub) {
        const int key = s0 + sub * 16 + l16;
        const _Float16* kr = kr_base + (size_t)key * EE;
        FragA bk0, bk1;
#pragma unroll
        for (int j = 0; j < 8; ++j) {
            const int d = ((j >> 2) << 4) + (half_ << 3) + ((j & 3) << 1);
            bk0.u[j] = *(const unsigned*)(kr + d);
            bk1.u[j] = *(const unsigned*)(kr + 32 + d);
        }
        v8f s = {};
        s = __builtin_amdgcn_wmma_f32_16x16x32_f16(false, aq0.h, false, bk0.h,
                                                   (short)0, s, false, false);
        s = __builtin_amdgcn_wmma_f32_16x16x32_f16(false, aq1.h, false, bk1.h,
                                                   (short)0, s, false, false);
        sc[sub] = s;
    }
#pragma unroll
    for (int r = 0; r < 8; ++r) {
        float s0v = sc[0][r];
        float s1v = sc[1][r];
        if (MASKED) {
            const int t = qt * 16 + half_ * 8 + r;
            if (s0 + l16 > t)      s0v = -__builtin_inff();
            if (s0 + 16 + l16 > t) s1v = -__builtin_inff();
        }
        float mx = fmaxf(s0v, s1v);
#pragma unroll
        for (int off = 1; off < 16; off <<= 1)
            mx = fmaxf(mx, __shfl_xor(mx, off, 32));
        const float nm    = fmaxf(row_max[r], mx);
        const float alpha = __expf(row_max[r] - nm);
        const float p0    = __expf(s0v - nm);
        const float p1    = __expf(s1v - nm);
        float ps = p0 + p1;
#pragma unroll
        for (int off = 1; off < 16; off <<= 1)
            ps += __shfl_xor(ps, off, 32);
        row_sum[r] = row_sum[r] * alpha + ps;
        row_max[r] = nm;
#pragma unroll
        for (int dt = 0; dt < 4; ++dt) o_acc[dt][r] *= alpha;
        pb[(half_ * 8 + r) * 32 + l16]      = (_Float16)p0;
        pb[(half_ * 8 + r) * 32 + 16 + l16] = (_Float16)p1;
    }
    // P (16x32) x V (32x64): rebuild A fragment from this wave's LDS tile
    // (same-wave RAW; DS ops are in-order, compiler inserts s_wait_dscnt).
    FragA ap;
#pragma unroll
    for (int j = 0; j < 8; ++j) {
        const int koff = ((j >> 2) << 4) + (half_ << 3) + ((j & 3) << 1);
        ap.u[j] = *(const unsigned*)&pb[l16 * 32 + koff];
    }
#pragma unroll
    for (int dt = 0; dt < 4; ++dt) {
        const int d = dt * 16 + l16;
        const _Float16* vrow = vt_base + (size_t)d * TT + s0;
        FragA bv;
#pragma unroll
        for (int j = 0; j < 8; ++j)
            bv.u[j] = *(const unsigned*)(vrow + ((j >> 2) << 4) + (half_ << 3) + ((j & 3) << 1));   // packed key pairs
        o_acc[dt] = __builtin_amdgcn_wmma_f32_16x16x32_f16(
            false, ap.h, false, bv.h, (short)0, o_acc[dt], false, false);
    }
}

__global__ __launch_bounds__(128)
void flash_attn_kernel(const _Float16* __restrict__ Q,
                       const _Float16* __restrict__ Km,
                       const _Float16* __restrict__ VT,
                       _Float16* __restrict__ O)
{
    __shared__ __align__(16) _Float16 pbuf[4][16 * 32];
    const int wv   = threadIdx.x >> 5;
    const int lane = threadIdx.x & 31;
    const int gw = blockIdx.x * 4 + wv;          // 0 .. B*H*(T/16)-1
    const int qt = gw & 127;                     // T/16 = 128 tiles
    const int hh = (gw >> 7) & 15;
    const int bb = gw >> 11;
    const int half_ = lane >> 4, l16 = lane & 15;

    const size_t headoff = (size_t)hh * HS;
    const size_t batch0  = (size_t)bb * TT;
    const _Float16* kr_base = Km + batch0 * EE + headoff;
    const _Float16* vt_base = VT + (size_t)(bb * HH + hh) * HS * TT;
    _Float16* pb = pbuf[wv];

    // Q fragments for contraction d = 0..63 (two k-steps of 32); scores were
    // pre-scaled by E^-0.5 in the Q projection GEMM epilogue.
    FragA aq0, aq1;
    {
        const _Float16* qr = Q + (batch0 + qt * 16 + l16) * EE + headoff;
#pragma unroll
        for (int j = 0; j < 8; ++j) {
            const int d = ((j >> 2) << 4) + (half_ << 3) + ((j & 3) << 1);
            aq0.u[j] = *(const unsigned*)(qr + d);
            aq1.u[j] = *(const unsigned*)(qr + 32 + d);
        }
    }

    float row_max[8], row_sum[8];
#pragma unroll
    for (int r = 0; r < 8; ++r) { row_max[r] = -__builtin_inff(); row_sum[r] = 0.f; }
    const v8f vzero = {};
    v8f o_acc[4];
#pragma unroll
    for (int dt = 0; dt < 4; ++dt) o_acc[dt] = vzero;

    const int tmax = qt * 16 + 15;
    const int last = (tmax >> 5) << 5;           // diagonal (masked) block
    for (int s0 = 0; s0 < last; s0 += 32)
        attn_block<false>(s0, qt, half_, l16, aq0, aq1, kr_base, vt_base, pb,
                          row_max, row_sum, o_acc);
    attn_block<true>(last, qt, half_, l16, aq0, aq1, kr_base, vt_base, pb,
                     row_max, row_sum, o_acc);

    // epilogue: divide by row sums, write f16
#pragma unroll
    for (int dt = 0; dt < 4; ++dt) {
#pragma unroll
        for (int r = 0; r < 8; ++r) {
            const int t = qt * 16 + half_ * 8 + r;
            O[(batch0 + t) * EE + headoff + dt * 16 + l16] =
                (_Float16)(o_acc[dt][r] / row_sum[r]);
        }
    }
}

// ---------------------------------------------------------------------------
extern "C" void kernel_launch(void* const* d_in, const int* in_sizes, int n_in,
                              void* d_out, int out_size, void* d_ws, size_t ws_size,
                              hipStream_t stream)
{
    (void)in_sizes; (void)n_in; (void)out_size; (void)ws_size;
    const float* x      = (const float*)d_in[0];
    const float* wq     = (const float*)d_in[1];
    const float* wk     = (const float*)d_in[2];
    const float* wvw    = (const float*)d_in[3];
    const float* wproj  = (const float*)d_in[4];
    const float* bproj  = (const float*)d_in[5];
    const float* w1     = (const float*)d_in[6];
    const float* b1     = (const float*)d_in[7];
    const float* w2     = (const float*)d_in[8];
    const float* b2     = (const float*)d_in[9];
    const float* ln1g   = (const float*)d_in[10];
    const float* ln1b   = (const float*)d_in[11];
    const float* ln2g   = (const float*)d_in[12];
    const float* ln2b   = (const float*)d_in[13];

    char* ws = (char*)d_ws;
    size_t off = 0;
    auto alloc = [&](size_t bytes) {
        size_t o = off; off += (bytes + 255) & ~(size_t)255; return o;
    };
    const size_t act16 = (size_t)MM * EE * sizeof(_Float16);   // 16 MB
    _Float16* hf16  = (_Float16*)(ws + alloc(act16));
    _Float16* qf    = (_Float16*)(ws + alloc(act16));
    _Float16* kf    = (_Float16*)(ws + alloc(act16));
    _Float16* vf    = (_Float16*)(ws + alloc(act16));
    _Float16* vtf   = (_Float16*)(ws + alloc(act16));
    _Float16* attO  = (_Float16*)(ws + alloc(act16));
    float*    x2    = (float*)(ws + alloc((size_t)MM * EE * sizeof(float)));
    _Float16* h2    = (_Float16*)(ws + alloc(act16));
    _Float16* ff1   = (_Float16*)(ws + alloc((size_t)MM * FFN_PAD * sizeof(_Float16)));
    _Float16* wqT   = (_Float16*)(ws + alloc((size_t)EE * EE * sizeof(_Float16)));
    _Float16* wkT   = (_Float16*)(ws + alloc((size_t)EE * EE * sizeof(_Float16)));
    _Float16* wvT   = (_Float16*)(ws + alloc((size_t)EE * EE * sizeof(_Float16)));
    _Float16* wpT   = (_Float16*)(ws + alloc((size_t)EE * EE * sizeof(_Float16)));
    _Float16* w1T   = (_Float16*)(ws + alloc((size_t)FFN_PAD * EE * sizeof(_Float16)));
    _Float16* w2T   = (_Float16*)(ws + alloc((size_t)EE * FFN_PAD * sizeof(_Float16)));

    // --- weight transposes (f32 -> f16, zero-padded) ---
    const int wgrid = (EE * EE + 255) / 256;
    wtrans_kernel<<<wgrid, 256, 0, stream>>>(wq,    wqT, EE, EE, EE, EE);
    wtrans_kernel<<<wgrid, 256, 0, stream>>>(wk,    wkT, EE, EE, EE, EE);
    wtrans_kernel<<<wgrid, 256, 0, stream>>>(wvw,   wvT, EE, EE, EE, EE);
    wtrans_kernel<<<wgrid, 256, 0, stream>>>(wproj, wpT, EE, EE, EE, EE);
    wtrans_kernel<<<(FFN_PAD * EE + 255) / 256, 256, 0, stream>>>(
        w1, w1T, EE, FFN_VALID, FFN_PAD, EE);
    wtrans_kernel<<<(EE * FFN_PAD + 255) / 256, 256, 0, stream>>>(
        w2, w2T, FFN_VALID, EE, EE, FFN_PAD);

    // --- LN1 ---
    ln_kernel<<<MM, 256, 0, stream>>>(x, ln1g, ln1b, hf16);

    // --- QKV projections (Q pre-scaled by E^-0.5 = 1/32) ---
    const int gemm_blocks = ((MM / 32) * (EE / 64) + 3) / 4;   // 1024
    wmma_gemm_kernel<true, false><<<gemm_blocks, 128, 0, stream>>>(
        hf16, wqT, nullptr, nullptr, qf, nullptr, MM, EE, EE, EE, EE, 0.03125f);
    wmma_gemm_kernel<true, false><<<gemm_blocks, 128, 0, stream>>>(
        hf16, wkT, nullptr, nullptr, kf, nullptr, MM, EE, EE, EE, EE, 1.0f);
    wmma_gemm_kernel<true, false><<<gemm_blocks, 128, 0, stream>>>(
        hf16, wvT, nullptr, nullptr, vf, nullptr, MM, EE, EE, EE, EE, 1.0f);

    // --- V transpose to [B,H,HS,T] for packed-pair B fragments ---
    vtrans_kernel<<<(MM * EE + 255) / 256, 256, 0, stream>>>(vf, vtf);

    // --- causal flash attention ---
    flash_attn_kernel<<<(BB * HH * (TT / 16)) / 4, 128, 0, stream>>>(
        qf, kf, vtf, attO);

    // --- output projection + residual -> x2 (f32) ---
    wmma_gemm_kernel<false, false><<<gemm_blocks, 128, 0, stream>>>(
        attO, wpT, bproj, x, nullptr, x2, MM, EE, EE, EE, EE, 1.0f);

    // --- LN2 ---
    ln_kernel<<<MM, 256, 0, stream>>>(x2, ln2g, ln2b, h2);

    // --- FFN1: relu(h2 @ w1 + b1), N padded 100->128 ---
    const int ffn1_blocks = ((MM / 32) * (FFN_PAD / 64) + 3) / 4;   // 128
    wmma_gemm_kernel<true, true><<<ffn1_blocks, 128, 0, stream>>>(
        h2, w1T, b1, nullptr, ff1, nullptr, MM, FFN_PAD, EE, FFN_VALID, FFN_PAD, 1.0f);

    // --- FFN2: x2 + ff1 @ w2 + b2 -> d_out (f32), K padded 100->128 ---
    wmma_gemm_kernel<false, false><<<gemm_blocks, 128, 0, stream>>>(
        ff1, w2T, b2, x2, nullptr, (float*)d_out, MM, EE, FFN_PAD, EE, EE, 1.0f);
}
